// SpatialCrossAttnBlock_62783831933255
// MI455X (gfx1250) — compile-verified, matrix operand order checked by reading, not run on hardware
//
#include <hip/hip_runtime.h>

#define DEVINL __device__ __forceinline__

typedef __bf16 bf16_t;
typedef __attribute__((ext_vector_type(16))) __bf16 v16bf;
typedef __attribute__((ext_vector_type(8)))  float    v8f;
typedef __attribute__((ext_vector_type(4)))  unsigned int v4u;

union Frag16 {
  v16bf bf;
  v4u   u[2];
  __bf16 e[16];
};

static constexpr int Bd = 4, Cd = 256, Td = 8, HWd = 1024;
static constexpr int NQ = Bd * Td * HWd;   // 32768 query positions
static constexpr int NR = Bd * HWd;        // 4096 reference positions

// Load a 16-element bf16 operand fragment from a row-major matrix row.
// A 16x32 / B 32x16 bf16 WMMA layout: lanes 0-15 use k-offsets {0..7,16..23},
// lanes 16-31 use {8..15,24..31}. col0 already includes (lane/16)*8.
DEVINL v16bf load_frag(const bf16_t* rowBase, int col0) {
  Frag16 f;
  f.u[0] = *(const v4u*)(rowBase + col0);
  f.u[1] = *(const v4u*)(rowBase + col0 + 16);
  return f.bf;
}

DEVINL v8f wmma_bf16(v16bf a, v16bf b, v8f c) {
  return __builtin_amdgcn_wmma_f32_16x16x32_bf16(false, a, false, b, (short)0, c,
                                                 false, false);
}

// Async DMA: 16 bytes global -> LDS, tracked by ASYNCcnt (CDNA5).
// Syntax per ISA 15.18.3: VDST = LDS byte address VGPR, then 64-bit global addr.
DEVINL void async_ld16(unsigned ldsAddr, const bf16_t* gaddr) {
  asm volatile("global_load_async_to_lds_b128 %0, %1, off"
               :: "v"(ldsAddr), "v"(gaddr) : "memory");
}

// ---------------- GroupNorm stats (two-stage, deterministic) ----------------

__global__ __launch_bounds__(256) void gn_partial_kernel(
    const float* __restrict__ x, long perSample, int blocksPer,
    float* __restrict__ partial)
{
  const int b   = blockIdx.x / blocksPer;
  const int blk = blockIdx.x % blocksPer;
  const float* xb = x + (size_t)b * perSample;
  float s = 0.f, ss = 0.f;
  for (long i = (long)blk * blockDim.x + threadIdx.x; i < perSample;
       i += (long)blocksPer * blockDim.x) {
    const float v = xb[i];
    s += v; ss += v * v;
  }
  __shared__ float r1[256], r2[256];
  const int tid = threadIdx.x;
  r1[tid] = s; r2[tid] = ss;
  __syncthreads();
  for (int o = 128; o > 0; o >>= 1) {
    if (tid < o) { r1[tid] += r1[tid + o]; r2[tid] += r2[tid + o]; }
    __syncthreads();
  }
  if (tid == 0) {
    partial[2 * ((size_t)b * blocksPer + blk) + 0] = r1[0];
    partial[2 * ((size_t)b * blocksPer + blk) + 1] = r2[0];
  }
}

__global__ __launch_bounds__(64) void gn_final_kernel(
    const float* __restrict__ partial, int blocksPer, float invN,
    float* __restrict__ stats)
{
  const int b = blockIdx.x;
  float s = 0.f, ss = 0.f;
  for (int i = threadIdx.x; i < blocksPer; i += 64) {
    s  += partial[2 * (b * blocksPer + i) + 0];
    ss += partial[2 * (b * blocksPer + i) + 1];
  }
  __shared__ float r1[64], r2[64];
  const int tid = threadIdx.x;
  r1[tid] = s; r2[tid] = ss;
  __syncthreads();
  for (int o = 32; o > 0; o >>= 1) {
    if (tid < o) { r1[tid] += r1[tid + o]; r2[tid] += r2[tid + o]; }
    __syncthreads();
  }
  if (tid == 0) {
    const float mu  = r1[0] * invN;
    const float var = r2[0] * invN - mu * mu;
    stats[2 * b + 0] = mu;
    stats[2 * b + 1] = rsqrtf(var + 1e-6f);
  }
}

// -------- Normalize + transpose [B,C,T,HW] f32 -> [pos, C] bf16 --------

__global__ __launch_bounds__(256) void gn_apply_kernel(
    const float* __restrict__ x, const float* __restrict__ stats,
    const float* __restrict__ gamma, const float* __restrict__ beta,
    bf16_t* __restrict__ out, int TT)
{
  __shared__ float tile[32][33];
  const int tx = threadIdx.x & 31;
  const int ty = threadIdx.x >> 5;          // 0..7
  const int posBase = blockIdx.x * 32;
  const int cBase   = blockIdx.y * 32;
  const int b  = posBase / (TT * 1024);
  const int t  = (posBase / 1024) % TT;
  const int hw = posBase & 1023;
  const float mu = stats[2 * b], rstd = stats[2 * b + 1];
#pragma unroll
  for (int i = 0; i < 4; ++i) {
    const int c = cBase + ty + i * 8;
    tile[ty + i * 8][tx] =
        x[(((size_t)b * Cd + c) * TT + t) * 1024 + hw + tx];
  }
  __syncthreads();
  const int c = cBase + tx;
  const float g  = gamma[c] * rstd;
  const float be = beta[c] - mu * g;        // (v-mu)*rstd*gamma + beta
#pragma unroll
  for (int i = 0; i < 4; ++i) {
    const int pr = ty + i * 8;
    out[(size_t)(posBase + pr) * Cd + c] = (bf16_t)(tile[tx][pr] * g + be);
  }
}

// -------------------------- weight f32 -> bf16 --------------------------

__global__ __launch_bounds__(256) void cvt_bf16_kernel(
    const float* __restrict__ in, bf16_t* __restrict__ out, int n)
{
  const int i = blockIdx.x * blockDim.x + threadIdx.x;
  if (i < n) out[i] = (bf16_t)in[i];
}

// -------- WMMA GEMM: [N,256] bf16 x Wmat[256,256]^T + bias --------
// MODE 0: row-major bf16 out.  MODE 1: transposed bf16 out (for V^T).
// MODE 2: f32 scatter to [B,C,T,HW] with residual add (output projection).

template <int MODE>
__global__ __launch_bounds__(256) void gemm256_kernel(
    const bf16_t* __restrict__ A, int nRows,
    const bf16_t* __restrict__ W, const float* __restrict__ bias, float scale,
    bf16_t* __restrict__ outBf, float* __restrict__ outF,
    const float* __restrict__ resid)
{
  const int lane = threadIdx.x & 31;
  const int wv   = threadIdx.x >> 5;
  const int half = lane >> 4;
  const int lm   = lane & 15;
  const int row0 = (blockIdx.x * 8 + wv) * 16;
  const int col0 = blockIdx.y * 16;
  if (row0 >= nRows) return;

  const bf16_t* aRow = A + (size_t)(row0 + lm) * 256;  // A lane m = lm
  const bf16_t* wRow = W + (size_t)(col0 + lm) * 256;  // B[k][n]=W[n][k], lane n = lm

  v8f acc = {0.f, 0.f, 0.f, 0.f, 0.f, 0.f, 0.f, 0.f};
#pragma unroll
  for (int kt = 0; kt < 8; ++kt) {
    const int c0 = kt * 32 + half * 8;
    acc = wmma_bf16(load_frag(aRow, c0), load_frag(wRow, c0), acc);
  }
  const float bcol = bias[col0 + lm];
#pragma unroll
  for (int r = 0; r < 8; ++r) {
    const int row = row0 + r + 8 * half;     // C/D: VGPR r -> row r (+8 for hi half)
    const float v = (acc[r] + bcol) * scale;
    if (MODE == 0) {
      outBf[(size_t)row * 256 + col0 + lm] = (bf16_t)v;
    } else if (MODE == 1) {                  // V^T: [b*256 + c][hw]
      const int bb = row >> 10, hw = row & 1023;
      outBf[((size_t)bb * 256 + (col0 + lm)) * 1024 + hw] = (bf16_t)v;
    } else {                                 // residual f32 scatter to NCTHW
      const int bb = row >> 13, t = (row >> 10) & 7, hw = row & 1023;
      const size_t idx = (((size_t)bb * 256 + (col0 + lm)) * 8 + t) * 1024 + hw;
      outF[idx] = resid[idx] + v;
    }
  }
}

// -------- Flash cross-attention --------
// One 16-query tile per wave over 1024 keys. S^T = K x Q^T so softmax rows are
// per-lane and P^T is directly the A-operand of the P x V WMMA. K and V^T chunk
// tiles are shared by all 8 waves of the block, so they are staged into LDS
// once per block with async global->LDS DMA (ASYNCcnt), double-buffered.

#define KPAD 264   // 256 + 8 bf16: 132-dword row stride -> conflict-free b128 reads
#define VPAD 40    // 32 + 8 bf16:  20-dword row stride  -> conflict-free b128 reads

__global__ __launch_bounds__(256) void attn_kernel(
    const bf16_t* __restrict__ Q, const bf16_t* __restrict__ Kf,
    const bf16_t* __restrict__ Vt, bf16_t* __restrict__ O)
{
  __shared__ __align__(16) bf16_t shK[2][32][KPAD];    // K chunk  [kp 32][c 256]
  __shared__ __align__(16) bf16_t shV[2][256][VPAD];   // V^T chunk [c 256][kp 32]

  const int lane = threadIdx.x & 31;
  const int wv   = threadIdx.x >> 5;
  const int half = lane >> 4;
  const int lm   = lane & 15;
  const int qRow0 = (blockIdx.x * 8 + wv) * 16;
  const int b = (blockIdx.x * 128) >> 13;            // whole block is one batch
  const bf16_t* Kb   = Kf + (size_t)b * 1024 * 256;  // [kp][c]
  const bf16_t* Vb   = Vt + (size_t)b * 256 * 1024;  // [c][kp]
  const bf16_t* qRow = Q + (size_t)(qRow0 + lm) * 256;

  // Stage chunk ch into buffer buf: 8 async b128 instructions per wave.
  auto stage = [&](int ch, int buf) {
    const int kp0 = ch * 32;
#pragma unroll
    for (int i = 0; i < 4; ++i) {                    // K: 1024 x 16B segments
      const int l = threadIdx.x + 256 * i;
      const int row = l >> 5, seg = l & 31;
      async_ld16((unsigned)(uintptr_t)&shK[buf][row][seg * 8],
                 Kb + (size_t)(kp0 + row) * 256 + seg * 8);
    }
#pragma unroll
    for (int i = 0; i < 4; ++i) {                    // V^T: 1024 x 16B segments
      const int l = threadIdx.x + 256 * i;
      const int c = l >> 2, seg = l & 3;
      async_ld16((unsigned)(uintptr_t)&shV[buf][c][seg * 8],
                 Vb + (size_t)c * 1024 + kp0 + seg * 8);
    }
  };

  float m_run = -3.0e30f, l_run = 0.f;
  v8f acc[16];
#pragma unroll
  for (int ct = 0; ct < 16; ++ct)
    acc[ct] = (v8f){0.f, 0.f, 0.f, 0.f, 0.f, 0.f, 0.f, 0.f};

  stage(0, 0);
  for (int ch = 0; ch < 32; ++ch) {                  // 32 key chunks of 32
    const int buf = ch & 1;
    if (ch + 1 < 32) {
      stage(ch + 1, buf ^ 1);                        // prefetch next chunk
      asm volatile("s_wait_asynccnt 0x8" ::: "memory");  // chunk ch complete
    } else {
      asm volatile("s_wait_asynccnt 0x0" ::: "memory");
    }
    __syncthreads();                                 // staging visible to all waves

    const bf16_t* kRow0 = &shK[buf][lm][0];
    const bf16_t* kRow1 = &shK[buf][16 + lm][0];
    v8f st0 = {0.f, 0.f, 0.f, 0.f, 0.f, 0.f, 0.f, 0.f};
    v8f st1 = {0.f, 0.f, 0.f, 0.f, 0.f, 0.f, 0.f, 0.f};
#pragma unroll
    for (int kt = 0; kt < 8; ++kt) {                 // C=256 in steps of 32
      const int c0 = kt * 32 + half * 8;
      const v16bf qb = load_frag(qRow, c0);          // B operand: lane n = q
      st0 = wmma_bf16(load_frag(kRow0, c0), qb, st0);
      st1 = wmma_bf16(load_frag(kRow1, c0), qb, st1);
    }
    // Per-lane softmax stats (lane holds query lm; partner lane other kp half)
    float mx = st0[0];
#pragma unroll
    for (int r = 1; r < 8; ++r) mx = fmaxf(mx, st0[r]);
#pragma unroll
    for (int r = 0; r < 8; ++r) mx = fmaxf(mx, st1[r]);
    mx = fmaxf(mx, __shfl_xor(mx, 16, 32));
    const float m_new = fmaxf(m_run, mx);
    float sum = 0.f;
    Frag16 pa;                                       // P^T == A-operand layout
#pragma unroll
    for (int r = 0; r < 8; ++r) {
      const float p = __expf(st0[r] - m_new); sum += p; pa.e[r] = (__bf16)p;
    }
#pragma unroll
    for (int r = 0; r < 8; ++r) {
      const float p = __expf(st1[r] - m_new); sum += p; pa.e[8 + r] = (__bf16)p;
    }
    sum += __shfl_xor(sum, 16, 32);
    const float alpha = __expf(m_run - m_new);
    l_run = l_run * alpha + sum;
    m_run = m_new;
    // Broadcast alpha[q] (held in lane q) to the C/D row layout, rescale O.
    float aq[8];
#pragma unroll
    for (int r = 0; r < 8; ++r) aq[r] = __shfl(alpha, r + 8 * half, 32);
#pragma unroll
    for (int ct = 0; ct < 16; ++ct)
#pragma unroll
      for (int r = 0; r < 8; ++r) acc[ct][r] *= aq[r];
    // O += P x V  (B operand: lane n = c reads V^T row contiguously in kp)
#pragma unroll
    for (int ct = 0; ct < 16; ++ct) {
      const bf16_t* vRow = &shV[buf][ct * 16 + lm][0];
      acc[ct] = wmma_bf16(pa.bf, load_frag(vRow, half * 8), acc[ct]);
    }
    __syncthreads();                                 // done reading buf before reuse
  }
  const float li = 1.0f / l_run;
  float linv[8];
#pragma unroll
  for (int r = 0; r < 8; ++r) linv[r] = __shfl(li, r + 8 * half, 32);
#pragma unroll
  for (int ct = 0; ct < 16; ++ct)
#pragma unroll
    for (int r = 0; r < 8; ++r) {
      const int row = qRow0 + r + 8 * half;
      O[(size_t)row * 256 + ct * 16 + lm] = (bf16_t)(acc[ct][r] * linv[r]);
    }
}

// ------------------------------- launch -------------------------------

extern "C" void kernel_launch(void* const* d_in, const int* in_sizes, int n_in,
                              void* d_out, int out_size, void* d_ws, size_t ws_size,
                              hipStream_t stream)
{
  (void)in_sizes; (void)n_in; (void)out_size; (void)ws_size;
  const float* x_mot = (const float*)d_in[0];
  const float* x_ref = (const float*)d_in[1];
  const float* gamma = (const float*)d_in[2];
  const float* beta  = (const float*)d_in[3];
  const float* Wq = (const float*)d_in[4];  const float* bq = (const float*)d_in[5];
  const float* Wk = (const float*)d_in[6];  const float* bk = (const float*)d_in[7];
  const float* Wv = (const float*)d_in[8];  const float* bv = (const float*)d_in[9];
  const float* Wo = (const float*)d_in[10]; const float* bo = (const float*)d_in[11];
  float* out = (float*)d_out;

  char* ws = (char*)d_ws;
  size_t off = 0;
  auto carve = [&](size_t bytes) -> char* {
    char* p = ws + off;
    off += (bytes + 255) & ~(size_t)255;
    return p;
  };
  float*  statsMot = (float*)carve(Bd * 2 * sizeof(float));
  float*  statsRef = (float*)carve(Bd * 2 * sizeof(float));
  float*  partMot  = (float*)carve(Bd * 64 * 2 * sizeof(float));
  float*  partRef  = (float*)carve(Bd * 16 * 2 * sizeof(float));
  bf16_t* WqB  = (bf16_t*)carve(65536 * sizeof(bf16_t));
  bf16_t* WkB  = (bf16_t*)carve(65536 * sizeof(bf16_t));
  bf16_t* WvB  = (bf16_t*)carve(65536 * sizeof(bf16_t));
  bf16_t* WoB  = (bf16_t*)carve(65536 * sizeof(bf16_t));
  bf16_t* Hmot = (bf16_t*)carve((size_t)NQ * 256 * sizeof(bf16_t));
  bf16_t* Href = (bf16_t*)carve((size_t)NR * 256 * sizeof(bf16_t));
  bf16_t* Qb   = (bf16_t*)carve((size_t)NQ * 256 * sizeof(bf16_t));
  bf16_t* Kfb  = (bf16_t*)carve((size_t)NR * 256 * sizeof(bf16_t));
  bf16_t* Vtb  = (bf16_t*)carve((size_t)NR * 256 * sizeof(bf16_t));
  bf16_t* Ob   = (bf16_t*)carve((size_t)NQ * 256 * sizeof(bf16_t));

  // Weights -> bf16
  cvt_bf16_kernel<<<256, 256, 0, stream>>>(Wq, WqB, 65536);
  cvt_bf16_kernel<<<256, 256, 0, stream>>>(Wk, WkB, 65536);
  cvt_bf16_kernel<<<256, 256, 0, stream>>>(Wv, WvB, 65536);
  cvt_bf16_kernel<<<256, 256, 0, stream>>>(Wo, WoB, 65536);

  // GroupNorm statistics (per sample)
  gn_partial_kernel<<<Bd * 64, 256, 0, stream>>>(x_mot, (long)Cd * Td * HWd, 64, partMot);
  gn_final_kernel<<<Bd, 64, 0, stream>>>(partMot, 64, 1.0f / (Cd * Td * HWd), statsMot);
  gn_partial_kernel<<<Bd * 16, 256, 0, stream>>>(x_ref, (long)Cd * HWd, 16, partRef);
  gn_final_kernel<<<Bd, 64, 0, stream>>>(partRef, 16, 1.0f / (Cd * HWd), statsRef);

  // Normalize + transpose to position-major bf16
  gn_apply_kernel<<<dim3(NQ / 32, Cd / 32), 256, 0, stream>>>(x_mot, statsMot, gamma, beta, Hmot, Td);
  gn_apply_kernel<<<dim3(NR / 32, Cd / 32), 256, 0, stream>>>(x_ref, statsRef, gamma, beta, Href, 1);

  // Q/K/V projections (WMMA). Q pre-scaled by C^-0.5 = 1/16.
  gemm256_kernel<0><<<dim3(NQ / 128, 16), 256, 0, stream>>>(Hmot, NQ, WqB, bq, 0.0625f, Qb, nullptr, nullptr);
  gemm256_kernel<0><<<dim3(NR / 128, 16), 256, 0, stream>>>(Href, NR, WkB, bk, 1.0f, Kfb, nullptr, nullptr);
  gemm256_kernel<1><<<dim3(NR / 128, 16), 256, 0, stream>>>(Href, NR, WvB, bv, 1.0f, Vtb, nullptr, nullptr);

  // Flash cross-attention: 2048 query tiles, 8 waves/block, LDS-staged K/V
  attn_kernel<<<NQ / 16 / 8, 256, 0, stream>>>(Qb, Kfb, Vtb, Ob);

  // Output projection + residual, scatter back to [B,C,T,H,W] f32
  gemm256_kernel<2><<<dim3(NQ / 128, 16), 256, 0, stream>>>(Ob, NQ, WoB, bo, 1.0f, nullptr, out, x_mot);
}
